// Net_41824391528637
// MI455X (gfx1250) — matrix operand order repaired and checked by reading, not compiled, not run on hardware
//
#include <hip/hip_runtime.h>
#include <math.h>

// MinCutPool GNN forward for MI455X (gfx1250, wave32).
// Heavy edge scatter is HBM/atomic bound (~1GB traffic @ 23.3TB/s);
// the N x 128 x 128 GEMMs are mapped onto v_wmma_f32_16x16x32_f16.

#define HID    128
#define KCLUS  4
#define OUTC   2
#define GBLOCK 12500
#define NGRAPH 8
#define EPSF   1e-15f

typedef __attribute__((ext_vector_type(16))) _Float16 v16h;
typedef __attribute__((ext_vector_type(8)))  float    v8f;

// ---------------- WMMA fragment loaders -----------------------------------
// A matrix 16x32 f16 layout (ISA 7.12.2): lane L -> row M = L%16,
// element i -> k = (i/8)*16 + (L/16)*8 + (i%8)   (two runs of 8 contiguous k)
__device__ __forceinline__ v16h load_A_frag(const float* __restrict__ row,
                                            int kbase, int lane) {
  v16h a;
  const int hi = (lane >> 4) << 3;  // 0 or 8
#pragma unroll
  for (int g = 0; g < 2; ++g) {
    const float4* p = (const float4*)(row + kbase + g * 16 + hi);
    const float4 f0 = p[0];
    const float4 f1 = p[1];
    a[g * 8 + 0] = (_Float16)f0.x; a[g * 8 + 1] = (_Float16)f0.y;
    a[g * 8 + 2] = (_Float16)f0.z; a[g * 8 + 3] = (_Float16)f0.w;
    a[g * 8 + 4] = (_Float16)f1.x; a[g * 8 + 5] = (_Float16)f1.y;
    a[g * 8 + 6] = (_Float16)f1.z; a[g * 8 + 7] = (_Float16)f1.w;
  }
  return a;
}

// B matrix 32x16 f16 layout: lane L -> col N = L%16,
// element i -> k = (i/8)*16 + (L/16)*8 + (i%8).   W is row-major [128][HID].
__device__ __forceinline__ v16h load_B_frag(const float* __restrict__ W,
                                            int kbase, int ntile, int lane) {
  v16h b;
  const int n  = ntile + (lane & 15);
  const int kh = kbase + ((lane >> 4) << 3);
#pragma unroll
  for (int i = 0; i < 16; ++i) {
    b[i] = (_Float16)W[(kh + (i & 7) + ((i >> 3) << 4)) * HID + n];
  }
  return b;
}

// ---------------- kernels --------------------------------------------------
__global__ void zero_ws_kernel(float* __restrict__ p, size_t n) {
  size_t i = (size_t)blockIdx.x * blockDim.x + threadIdx.x;
  const size_t stride = (size_t)gridDim.x * blockDim.x;
  for (; i < n; i += stride) p[i] = 0.0f;
}

// agg[dst] += x[src] (128 f32), deg[dst] += 1.  One wave per edge.
__global__ __launch_bounds__(256) void edge_scatter_kernel(
    const float* __restrict__ x, const int* __restrict__ src,
    const int* __restrict__ dst, float* __restrict__ agg,
    float* __restrict__ deg, int E) {
  const int lane = threadIdx.x & 31;
  const int e = blockIdx.x * 8 + (threadIdx.x >> 5);
  if (e >= E) return;
  const int si = src[e];
  const int di = dst[e];
  const float4 v = *(const float4*)(x + (size_t)si * HID + lane * 4);
  float* dp = agg + (size_t)di * HID + lane * 4;
  atomicAdd(dp + 0, v.x);
  atomicAdd(dp + 1, v.y);
  atomicAdd(dp + 2, v.z);
  atomicAdd(dp + 3, v.w);
  if (lane == 0) atomicAdd(deg + di, 1.0f);
}

// h = relu(agg@W_rel1 + x@W_root1 + b_rel1) via WMMA (128 rows / block),
// then softmax s = sm(h@W_pool+b_pool), pooled features, ss, den partials.
__global__ __launch_bounds__(256) void fused_node_kernel(
    const float* __restrict__ x, const float* __restrict__ agg,
    const float* __restrict__ deg,
    const float* __restrict__ W_root1, const float* __restrict__ W_rel1,
    const float* __restrict__ b_rel1,
    const float* __restrict__ W_pool, const float* __restrict__ b_pool,
    float* __restrict__ s_dout, float* __restrict__ s_buf,
    float* __restrict__ pool_g, float* __restrict__ ss_g,
    float* __restrict__ den_g, int N) {
  __shared__ float h_l[128 * 132];   // padded stride vs LDS banks
  __shared__ float s_l[128 * 4];

  const int tid = threadIdx.x;
  const int lane = tid & 31;
  const int wave = tid >> 5;
  const int blockRow0 = blockIdx.x * 128;

  {  // ---- WMMA phase: each wave computes a 16x128 tile of h -------------
    int rr = blockRow0 + wave * 16 + (lane & 15);
    if (rr > N - 1) rr = N - 1;  // clamp; invalid rows masked later
    const float* aggrow = agg + (size_t)rr * HID;
    const float* xrow   = x   + (size_t)rr * HID;

    v8f acc[8] = {};
#pragma unroll
    for (int ks = 0; ks < 4; ++ks) {
      const v16h a = load_A_frag(aggrow, ks * 32, lane);
#pragma unroll
      for (int nt = 0; nt < 8; ++nt) {
        const v16h b = load_B_frag(W_rel1, ks * 32, nt * 16, lane);
        acc[nt] = __builtin_amdgcn_wmma_f32_16x16x32_f16(
            false, a, false, b, (short)0, acc[nt], false, false);
      }
    }
#pragma unroll
    for (int ks = 0; ks < 4; ++ks) {
      const v16h a = load_A_frag(xrow, ks * 32, lane);
#pragma unroll
      for (int nt = 0; nt < 8; ++nt) {
        const v16h b = load_B_frag(W_root1, ks * 32, nt * 16, lane);
        acc[nt] = __builtin_amdgcn_wmma_f32_16x16x32_f16(
            false, a, false, b, (short)0, acc[nt], false, false);
      }
    }
    // C/D layout: lane<16 -> M=v, N=lane ; lane>=16 -> M=8+v, N=lane-16
    const int mbase = wave * 16 + ((lane >> 4) << 3);
    const int ncol  = lane & 15;
#pragma unroll
    for (int nt = 0; nt < 8; ++nt) {
      const int col = nt * 16 + ncol;
      const float bias = b_rel1[col];
#pragma unroll
      for (int v = 0; v < 8; ++v) {
        h_l[(mbase + v) * 132 + col] = fmaxf(acc[nt][v] + bias, 0.0f);
      }
    }
  }
  __syncthreads();

  // ---- softmax over K=4 clusters ---------------------------------------
  if (tid < 128) {
    const int node = blockRow0 + tid;
    if (node < N) {
      float z0 = b_pool[0], z1 = b_pool[1], z2 = b_pool[2], z3 = b_pool[3];
      const float* hr = &h_l[tid * 132];
      for (int c = 0; c < HID; ++c) {
        const float hv = hr[c];
        const float4 wp = *(const float4*)(W_pool + c * 4);
        z0 += hv * wp.x; z1 += hv * wp.y; z2 += hv * wp.z; z3 += hv * wp.w;
      }
      const float mx = fmaxf(fmaxf(z0, z1), fmaxf(z2, z3));
      float e0 = expf(z0 - mx), e1 = expf(z1 - mx);
      float e2 = expf(z2 - mx), e3 = expf(z3 - mx);
      const float inv = 1.0f / (e0 + e1 + e2 + e3);
      e0 *= inv; e1 *= inv; e2 *= inv; e3 *= inv;
      s_l[tid * 4 + 0] = e0; s_l[tid * 4 + 1] = e1;
      s_l[tid * 4 + 2] = e2; s_l[tid * 4 + 3] = e3;
      s_dout[(size_t)node * 4 + 0] = e0;
      s_dout[(size_t)node * 4 + 1] = e1;
      s_dout[(size_t)node * 4 + 2] = e2;
      s_dout[(size_t)node * 4 + 3] = e3;
      float4 sv; sv.x = e0; sv.y = e1; sv.z = e2; sv.w = e3;
      *(float4*)(s_buf + (size_t)node * 4) = sv;
    } else {
      s_l[tid * 4 + 0] = 0.0f; s_l[tid * 4 + 1] = 0.0f;
      s_l[tid * 4 + 2] = 0.0f; s_l[tid * 4 + 3] = 0.0f;
    }
  }
  __syncthreads();

  const int bmin = blockRow0 / GBLOCK;  // block spans at most 2 graphs

  // ---- pooled features: out[b][k][c] += s[k]*h[c] ----------------------
  {
    const int c = tid & 127;
    const int half = tid >> 7;
    float a0[4] = {0, 0, 0, 0}, a1[4] = {0, 0, 0, 0};
    bool has1 = false;
    for (int r = half * 64; r < half * 64 + 64; ++r) {
      const int node = blockRow0 + r;
      if (node >= N) break;
      const int bs = node / GBLOCK - bmin;
      const float hv = h_l[r * 132 + c];
      const float* sp = &s_l[r * 4];
      if (bs == 0) {
#pragma unroll
        for (int k = 0; k < 4; ++k) a0[k] += sp[k] * hv;
      } else {
        has1 = true;
#pragma unroll
        for (int k = 0; k < 4; ++k) a1[k] += sp[k] * hv;
      }
    }
#pragma unroll
    for (int k = 0; k < 4; ++k)
      atomicAdd(&pool_g[bmin * (KCLUS * HID) + k * HID + c], a0[k]);
    if (has1) {
#pragma unroll
      for (int k = 0; k < 4; ++k)
        atomicAdd(&pool_g[(bmin + 1) * (KCLUS * HID) + k * HID + c], a1[k]);
    }
  }

  // ---- ss[b][k][l] += s[k]*s[l] ----------------------------------------
  if (tid < 32) {
    const int kl = tid & 15;
    const int half = tid >> 4;
    const int kk = kl >> 2, ll = kl & 3;
    float a0 = 0.0f, a1 = 0.0f;
    bool has1 = false;
    for (int r = half * 64; r < half * 64 + 64; ++r) {
      const int node = blockRow0 + r;
      if (node >= N) break;
      const int bs = node / GBLOCK - bmin;
      const float v = s_l[r * 4 + kk] * s_l[r * 4 + ll];
      if (bs == 0) a0 += v; else { has1 = true; a1 += v; }
    }
    atomicAdd(&ss_g[bmin * 16 + kl], a0);
    if (has1) atomicAdd(&ss_g[(bmin + 1) * 16 + kl], a1);
  } else if (tid < 34) {  // ---- den[b] += deg*sum(s^2) --------------------
    const int half = tid - 32;
    float a0 = 0.0f, a1 = 0.0f;
    bool has1 = false;
    for (int r = half * 64; r < half * 64 + 64; ++r) {
      const int node = blockRow0 + r;
      if (node >= N) break;
      const int bs = node / GBLOCK - bmin;
      const float* sp = &s_l[r * 4];
      const float sum2 = sp[0]*sp[0] + sp[1]*sp[1] + sp[2]*sp[2] + sp[3]*sp[3];
      const float v = deg[node] * sum2;
      if (bs == 0) a0 += v; else { has1 = true; a1 += v; }
    }
    atomicAdd(&den_g[bmin], a0);
    if (has1) atomicAdd(&den_g[bmin + 1], a1);
  }
}

// out_adj[b][k][l] += s[dst][k]*s[src][l] via per-block LDS partials.
__global__ __launch_bounds__(256) void edge_adj_kernel(
    const int* __restrict__ src, const int* __restrict__ dst,
    const float* __restrict__ s_buf, float* __restrict__ adj_g, int E) {
  __shared__ float adj_l[NGRAPH * 16];
  const int tid = threadIdx.x;
  if (tid < NGRAPH * 16) adj_l[tid] = 0.0f;
  __syncthreads();
  const int e = blockIdx.x * 256 + tid;
  if (e < E) {
    const int si = src[e], di = dst[e];
    const float4 sd  = *(const float4*)(s_buf + (size_t)di * 4);
    const float4 ssv = *(const float4*)(s_buf + (size_t)si * 4);
    const int b = di / GBLOCK;
    float dv[4] = {sd.x, sd.y, sd.z, sd.w};
    float sv[4] = {ssv.x, ssv.y, ssv.z, ssv.w};
    float* base = &adj_l[b * 16];
#pragma unroll
    for (int k = 0; k < 4; ++k)
#pragma unroll
      for (int l = 0; l < 4; ++l)
        atomicAdd(&base[k * 4 + l], dv[k] * sv[l]);
  }
  __syncthreads();
  if (tid < NGRAPH * 16) {
    const float v = adj_l[tid];
    if (v != 0.0f) atomicAdd(&adj_g[tid], v);
  }
}

// Losses, adjacency fix/normalize, dense conv, MLP head. One block.
__global__ __launch_bounds__(256) void readout_kernel(
    const float* __restrict__ pool_g, const float* __restrict__ adj_g,
    const float* __restrict__ ss_g, const float* __restrict__ den_g,
    const float* __restrict__ W_rel3, const float* __restrict__ b_rel3,
    const float* __restrict__ W_root3,
    const float* __restrict__ W_lin1, const float* __restrict__ b_lin1,
    const float* __restrict__ W_lin2, const float* __restrict__ b_lin2,
    float* __restrict__ out, float* __restrict__ adj_out) {
  __shared__ float adjn[NGRAPH * 16];
  __shared__ float x2a[NGRAPH * KCLUS * HID];
  __shared__ float yv[NGRAPH * KCLUS * HID];
  __shared__ float mv[NGRAPH * HID];
  __shared__ float rv[NGRAPH * HID];
  __shared__ float mcs[NGRAPH], osv[NGRAPH];
  const int tid = threadIdx.x;

  if (tid < NGRAPH) {
    const float* A = &adj_g[tid * 16];
    const float tr = A[0] + A[5] + A[10] + A[15];
    mcs[tid] = -(tr / den_g[tid]);
    const float* S = &ss_g[tid * 16];
    float nrm = 0.0f;
    for (int i = 0; i < 16; ++i) nrm += S[i] * S[i];
    nrm = sqrtf(nrm);
    float od = 0.0f;
    for (int k = 0; k < 4; ++k)
      for (int l = 0; l < 4; ++l) {
        const float v = S[k * 4 + l] / nrm - ((k == l) ? 0.5f : 0.0f);
        od += v * v;
      }
    osv[tid] = sqrtf(od);
    float Az[16];
    for (int k = 0; k < 4; ++k)
      for (int l = 0; l < 4; ++l)
        Az[k * 4 + l] = (k == l) ? 0.0f : A[k * 4 + l];
    float dd[4];
    for (int k = 0; k < 4; ++k)
      dd[k] = sqrtf(Az[k*4+0] + Az[k*4+1] + Az[k*4+2] + Az[k*4+3]) + EPSF;
    for (int k = 0; k < 4; ++k)
      for (int l = 0; l < 4; ++l) {
        const float v = Az[k * 4 + l] / dd[l] / dd[k];
        adjn[tid * 16 + k * 4 + l] = v;
        adj_out[tid * 16 + k * 4 + l] = v;
      }
  }
  __syncthreads();
  if (tid == 0) {
    float a = 0.0f, b = 0.0f;
    for (int i = 0; i < NGRAPH; ++i) { a += mcs[i]; b += osv[i]; }
    out[NGRAPH * OUTC + 0] = a / (float)NGRAPH;  // mc_loss  (idx 16)
    out[NGRAPH * OUTC + 1] = b / (float)NGRAPH;  // o_loss   (idx 17)
  }
  // x2a = adj_norm @ pooled
  for (int idx = tid; idx < NGRAPH * KCLUS * HID; idx += 256) {
    const int b = idx >> 9, k = (idx >> 7) & 3, c = idx & 127;
    float acc = 0.0f;
#pragma unroll
    for (int l = 0; l < 4; ++l)
      acc += adjn[b * 16 + k * 4 + l] * pool_g[b * 512 + l * 128 + c];
    x2a[idx] = acc;
  }
  __syncthreads();
  // y = x2a@W_rel3 + b_rel3 + pooled@W_root3
  for (int idx = tid; idx < NGRAPH * KCLUS * HID; idx += 256) {
    const int b = idx >> 9, k = (idx >> 7) & 3, c = idx & 127;
    float acc = b_rel3[c];
    const float* xr = &x2a[b * 512 + k * 128];
    const float* pr = &pool_g[b * 512 + k * 128];
    for (int cc = 0; cc < HID; ++cc)
      acc += xr[cc] * W_rel3[cc * HID + c] + pr[cc] * W_root3[cc * HID + c];
    yv[idx] = acc;
  }
  __syncthreads();
  for (int idx = tid; idx < NGRAPH * HID; idx += 256) {
    const int b = idx >> 7, c = idx & 127;
    mv[idx] = 0.25f * (yv[b * 512 + c] + yv[b * 512 + 128 + c] +
                       yv[b * 512 + 256 + c] + yv[b * 512 + 384 + c]);
  }
  __syncthreads();
  for (int idx = tid; idx < NGRAPH * HID; idx += 256) {
    const int b = idx >> 7, c = idx & 127;
    float acc = b_lin1[c];
    const float* mr = &mv[b * 128];
    for (int cc = 0; cc < HID; ++cc) acc += mr[cc] * W_lin1[cc * HID + c];
    rv[idx] = fmaxf(acc, 0.0f);
  }
  __syncthreads();
  if (tid < NGRAPH) {
    float l0 = b_lin2[0], l1 = b_lin2[1];
    const float* rr = &rv[tid * 128];
    for (int cc = 0; cc < HID; ++cc) {
      l0 += rr[cc] * W_lin2[cc * 2 + 0];
      l1 += rr[cc] * W_lin2[cc * 2 + 1];
    }
    const float mx = fmaxf(l0, l1);
    const float lse = mx + logf(expf(l0 - mx) + expf(l1 - mx));
    out[tid * 2 + 0] = l0 - lse;
    out[tid * 2 + 1] = l1 - lse;
  }
}

// ---------------- host launcher -------------------------------------------
extern "C" void kernel_launch(void* const* d_in, const int* in_sizes, int n_in,
                              void* d_out, int out_size, void* d_ws,
                              size_t ws_size, hipStream_t stream) {
  const float* x       = (const float*)d_in[0];
  const int*   ei      = (const int*)d_in[1];
  // d_in[2] = batch (recomputed as node/GBLOCK on device)
  const float* W_root1 = (const float*)d_in[3];
  const float* W_rel1  = (const float*)d_in[4];
  const float* b_rel1  = (const float*)d_in[5];
  const float* W_pool  = (const float*)d_in[6];
  const float* b_pool  = (const float*)d_in[7];
  const float* W_rel3  = (const float*)d_in[8];
  const float* b_rel3  = (const float*)d_in[9];
  const float* W_root3 = (const float*)d_in[10];
  const float* W_lin1  = (const float*)d_in[11];
  const float* b_lin1  = (const float*)d_in[12];
  const float* W_lin2  = (const float*)d_in[13];
  const float* b_lin2  = (const float*)d_in[14];

  const int N = in_sizes[0] / HID;
  const int E = in_sizes[1] / 2;
  const int* src = ei;
  const int* dst = ei + E;

  // d_out layout: logits[B*2] | mc_loss | o_loss | s[N*4] | out_adj[B*16]
  float* out     = (float*)d_out;
  float* s_dout  = out + NGRAPH * OUTC + 2;
  float* adj_out = s_dout + (size_t)N * KCLUS;

  float* ws = (float*)d_ws;
  size_t off = 0;
  float* agg    = ws + off; off += (size_t)N * HID;
  float* deg    = ws + off; off += (size_t)N;
  float* s_buf  = ws + off; off += (size_t)N * KCLUS;   // 16B-aligned copy of s
  float* pool_g = ws + off; off += NGRAPH * KCLUS * HID;
  float* adj_g  = ws + off; off += NGRAPH * 16;
  float* ss_g   = ws + off; off += NGRAPH * 16;
  float* den_g  = ws + off; off += NGRAPH;
  const size_t zn = off;

  zero_ws_kernel<<<4096, 256, 0, stream>>>(ws, zn);
  edge_scatter_kernel<<<(E + 7) / 8, 256, 0, stream>>>(x, src, dst, agg, deg, E);
  fused_node_kernel<<<(N + 127) / 128, 256, 0, stream>>>(
      x, agg, deg, W_root1, W_rel1, b_rel1, W_pool, b_pool,
      s_dout, s_buf, pool_g, ss_g, den_g, N);
  edge_adj_kernel<<<(E + 255) / 256, 256, 0, stream>>>(src, dst, s_buf, adj_g, E);
  readout_kernel<<<1, 256, 0, stream>>>(
      pool_g, adj_g, ss_g, den_g, W_rel3, b_rel3, W_root3,
      W_lin1, b_lin1, W_lin2, b_lin2, out, adj_out);
}
